// FCbinWAInt_31653908971649
// MI455X (gfx1250) — 8-bit matrix operand order repaired and checked by reading, compiled, not run on hardware
//
#include <hip/hip_runtime.h>
#include <hip/hip_bf16.h>

#define BATCH   32768
#define H       512
#define DPAD    832
#define KT1     8
#define NT1     32
#define KT2     13
#define TSTEPS  15

typedef int v8i __attribute__((ext_vector_type(8)));

#define WMMA_IU8(A, B, C) \
  __builtin_amdgcn_wmma_i32_16x16x64_iu8(false, (A), true, (B), (C), false, false)

#define OFF_W1N   ((size_t)0)
#define OFF_W1T   ((size_t)262144)
#define OFF_W0N   ((size_t)524288)
#define OFF_W0T   ((size_t)557056)
#define OFF_W2T   ((size_t)565248)
#define OFF_DU8   ((size_t)991232)
#define OFF_D2    ((size_t)(991232 + (size_t)BATCH * DPAD))

__device__ __forceinline__ int a_k0(int v, int lane) {
  return ((v & 1) << 2) + (((v >> 1) & 1) << 4) + ((v >> 2) << 5) + ((lane >> 4) << 3);
}

__device__ __forceinline__ v8i load_bfrag(const int4* __restrict__ p,
                                          int nt, int kt, int KT, int lane) {
  const int4* q = p + (((size_t)(nt * KT + kt) * 32 + lane) * 2);
  int4 a = q[0], b = q[1];
  v8i r;
  r[0] = a.x; r[1] = a.y; r[2] = a.z; r[3] = a.w;
  r[4] = b.x; r[5] = b.y; r[6] = b.z; r[7] = b.w;
  return r;
}

__device__ __forceinline__ v8i make_afrag(const unsigned char* s, int rowBase,
                                          int ld, int ktOff, int lane) {
  const unsigned char* rowp = s + (size_t)(rowBase + (lane & 15)) * ld + ktOff;
  v8i r;
#pragma unroll
  for (int v = 0; v < 8; ++v) {
    unsigned x = *(const unsigned*)(rowp + a_k0(v, lane));
    r[v] = (int)((x >> 6) & 0x01010101u);
  }
  return r;
}

__device__ __forceinline__ int clamp127(int v) {
  return v < 0 ? 0 : (v > 127 ? 127 : v);
}

__global__ void pack_b_kernel(int* __restrict__ dst, const float* __restrict__ src,
                              int NT, int KT, int n_valid, int k_valid, int ld,
                              int transpose) {
  int idx = blockIdx.x * blockDim.x + threadIdx.x;
  int total = NT * KT * 256;
  if (idx >= total) return;
  int v    = idx & 7;
  int lane = (idx >> 3) & 31;
  int tile = idx >> 8;
  int kt   = tile % KT;
  int nt   = tile / KT;
  int n    = nt * 16 + (lane & 15);
  unsigned d = 0;
#pragma unroll
  for (int j = 0; j < 4; ++j) {
    int k = kt * 64 + ((v & 1) << 2) + (((v >> 1) & 1) << 4) + j + ((v >> 2) << 5) + ((lane >> 4) << 3);
    int val = 0;
    if (k < k_valid && n < n_valid) {
      float f = transpose ? src[(size_t)n * ld + k] : src[(size_t)k * ld + n];
      val = (int)f;
    }
    d |= ((unsigned)(val & 0xFF)) << (8 * j);
  }
  dst[idx] = (int)d;
}

__global__ void pack_data_kernel(unsigned char* __restrict__ dst,
                                 const float* __restrict__ data) {
  int d = blockIdx.x * blockDim.x + threadIdx.x;
  int total = BATCH * (DPAD / 4);
  if (d >= total) return;
  int r  = d / (DPAD / 4);
  int c4 = (d % (DPAD / 4)) * 4;
  unsigned w = 0;
#pragma unroll
  for (int j = 0; j < 4; ++j) {
    int c = c4 + j;
    unsigned val = (c < 784) ? (unsigned)(int)data[(size_t)r * 784 + c] : 0u;
    w |= (val & 0xFF) << (8 * j);
  }
  ((unsigned*)dst)[d] = w;
}

__global__ void __launch_bounds__(256) d2_kernel(const unsigned char* __restrict__ du8,
                                                 const int4* __restrict__ W2t,
                                                 short* __restrict__ D2) {
  int wave = (blockIdx.x * 256 + threadIdx.x) >> 5;
  int lane = threadIdx.x & 31;
  int nt      = wave & (NT1 - 1);
  int rowTile = wave >> 5;
  int rows    = rowTile * 16;

  const unsigned char* rowp = du8 + (size_t)(rows + (lane & 15)) * DPAD;
  v8i acc = {};
  for (int kt = 0; kt < KT2; ++kt) {
    v8i A;
#pragma unroll
    for (int v = 0; v < 8; ++v)
      A[v] = (int)*(const unsigned*)(rowp + kt * 64 + a_k0(v, lane));
    v8i Bf = load_bfrag(W2t, nt, kt, KT2, lane);
    acc = WMMA_IU8(A, Bf, acc);
  }
  int cm = (lane >> 4) * 8, cn = lane & 15;
#pragma unroll
  for (int g = 0; g < 8; ++g)
    D2[(size_t)(rows + g + cm) * H + nt * 16 + cn] = (short)acc[g];
}

__global__ void __launch_bounds__(256) relax_kernel(const short* __restrict__ D2,
                                                    const int4* __restrict__ W1n,
                                                    const int4* __restrict__ W1t,
                                                    const int4* __restrict__ W0n,
                                                    const int4* __restrict__ W0t,
                                                    float* __restrict__ out) {
  extern __shared__ unsigned char smem[];
  unsigned char* s1b = smem;
  unsigned char* s2b = smem + 131072;
  unsigned char* s0b = smem + 262144;

  const int tid  = threadIdx.x;
  const int wave = tid >> 5, lane = tid & 31;
  const int lRow = wave * 16;
  const int gRow = blockIdx.x * 128 + lRow;

  for (int buf = 0; buf < 2; ++buf) {
    unsigned* p1 = (unsigned*)(s1b + buf * 65536 + lRow * H);
    unsigned* p2 = (unsigned*)(s2b + buf * 65536 + lRow * H);
    for (int i = lane; i < 16 * H / 4; i += 32) { p1[i] = 0u; p2[i] = 0u; }
    unsigned* p0 = (unsigned*)(s0b + buf * 8192 + lRow * 64);
    for (int i = lane; i < 16 * 64 / 4; i += 32) p0[i] = 0u;
  }

  const int cm = (lane >> 4) * 8;
  const int cn = lane & 15;

  int cur = 0;
  for (int t = 0; t < TSTEPS; ++t) {
    unsigned char* s1c = s1b + cur * 65536;
    unsigned char* s2c = s2b + cur * 65536;
    unsigned char* s0c = s0b + cur * 8192;
    unsigned char* s1n = s1b + (cur ^ 1) * 65536;
    unsigned char* s2n = s2b + (cur ^ 1) * 65536;
    unsigned char* s0n = s0b + (cur ^ 1) * 8192;

    v8i A1[KT1], A2[KT1], A0;
#pragma unroll
    for (int kt = 0; kt < KT1; ++kt) {
      A1[kt] = make_afrag(s1c, lRow, H, kt * 64, lane);
      A2[kt] = make_afrag(s2c, lRow, H, kt * 64, lane);
    }
    A0 = make_afrag(s0c, lRow, 64, 0, lane);

    for (int nt = 0; nt < NT1; ++nt) {
      v8i acc1 = {};
      v8i acc2 = {};
#pragma unroll
      for (int kt = 0; kt < KT1; ++kt) {
        v8i bn = load_bfrag(W1n, nt, kt, KT1, lane);
        acc2 = WMMA_IU8(A1[kt], bn, acc2);
        v8i bt = load_bfrag(W1t, nt, kt, KT1, lane);
        acc1 = WMMA_IU8(A2[kt], bt, acc1);
      }
      {
        v8i b0f = load_bfrag(W0n, nt, 0, 1, lane);
        acc1 = WMMA_IU8(A0, b0f, acc1);
      }
#pragma unroll
      for (int g = 0; g < 8; ++g) {
        int r  = lRow + g + cm;
        int c  = nt * 16 + cn;
        int gr = gRow + g + cm;
        int pre2 = acc2[g] + (int)D2[(size_t)gr * H + c];
        int sn2  = clamp127((int)s2c[(size_t)r * H + c] + pre2 >= 0
                              ? ((int)s2c[(size_t)r * H + c] + pre2) / 2
                              : ((int)s2c[(size_t)r * H + c] + pre2) / 2);
        s2n[(size_t)r * H + c] = (unsigned char)sn2;
        int pre1 = acc1[g];
        int v1   = (int)s1c[(size_t)r * H + c] + pre1;
        s1n[(size_t)r * H + c] = (unsigned char)clamp127(v1 / 2);
      }
    }

    {
      v8i acc0 = {};
#pragma unroll
      for (int kt = 0; kt < KT1; ++kt) {
        v8i bw = load_bfrag(W0t, 0, kt, KT1, lane);
        acc0 = WMMA_IU8(A1[kt], bw, acc0);
      }
#pragma unroll
      for (int g = 0; g < 8; ++g) {
        int r = lRow + g + cm;
        int sn0 = 0;
        if (cn < 10) {
          int v0 = (int)s0c[r * 64 + cn] + acc0[g];
          sn0 = clamp127(v0 / 2);
        }
        s0n[r * 64 + cn] = (unsigned char)sn0;
      }
    }
    cur ^= 1;
  }

  unsigned char* s1c = s1b + cur * 65536;
  unsigned char* s2c = s2b + cur * 65536;
  unsigned char* s0c = s0b + cur * 8192;
  for (int i = lane; i < 16 * 10; i += 32) {
    int r = i / 10, n = i % 10;
    out[(size_t)(gRow + r) * 10 + n] = (float)s0c[(lRow + r) * 64 + n];
  }
  const size_t off1 = (size_t)BATCH * 10;
  const size_t off2 = off1 + (size_t)BATCH * H;
  for (int i = lane; i < 16 * H; i += 32) {
    int r = i >> 9, n = i & (H - 1);
    out[off1 + (size_t)(gRow + r) * H + n] = (float)s1c[(size_t)(lRow + r) * H + n];
    out[off2 + (size_t)(gRow + r) * H + n] = (float)s2c[(size_t)(lRow + r) * H + n];
  }
}

extern "C" void kernel_launch(void* const* d_in, const int* in_sizes, int n_in,
                              void* d_out, int out_size, void* d_ws, size_t ws_size,
                              hipStream_t stream) {
  const float* data = (const float*)d_in[3];
  const float* W0   = (const float*)d_in[4];
  const float* W1   = (const float*)d_in[5];
  const float* W2   = (const float*)d_in[6];

  char* ws = (char*)d_ws;
  int*  W1n = (int*)(ws + OFF_W1N);
  int*  W1t = (int*)(ws + OFF_W1T);
  int*  W0n = (int*)(ws + OFF_W0N);
  int*  W0t = (int*)(ws + OFF_W0T);
  int*  W2t = (int*)(ws + OFF_W2T);
  unsigned char* du8 = (unsigned char*)(ws + OFF_DU8);
  short* D2 = (short*)(ws + OFF_D2);

  pack_b_kernel<<<(NT1 * KT1 * 256 + 255) / 256, 256, 0, stream>>>(W1n, W1, NT1, KT1, 512, 512, 512, 0);
  pack_b_kernel<<<(NT1 * KT1 * 256 + 255) / 256, 256, 0, stream>>>(W1t, W1, NT1, KT1, 512, 512, 512, 1);
  pack_b_kernel<<<(NT1 * 1   * 256 + 255) / 256, 256, 0, stream>>>(W0n, W0, NT1, 1,   512, 10,  512, 0);
  pack_b_kernel<<<(1   * KT1 * 256 + 255) / 256, 256, 0, stream>>>(W0t, W0, 1,   KT1, 10,  512, 512, 1);
  pack_b_kernel<<<(NT1 * KT2 * 256 + 255) / 256, 256, 0, stream>>>(W2t, W2, NT1, KT2, 512, 784, 784, 1);

  pack_data_kernel<<<(BATCH * (DPAD / 4) + 255) / 256, 256, 0, stream>>>(du8, data);

  d2_kernel<<<(BATCH / 16) * NT1 / 8, 256, 0, stream>>>(du8, (const int4*)W2t, D2);

  const size_t smem_bytes = 131072 + 131072 + 16384;
  (void)hipFuncSetAttribute((const void*)relax_kernel,
                            hipFuncAttributeMaxDynamicSharedMemorySize,
                            (int)smem_bytes);
  relax_kernel<<<BATCH / 128, 256, smem_bytes, stream>>>(
      D2, (const int4*)W1n, (const int4*)W1t, (const int4*)W0n, (const int4*)W0t,
      (float*)d_out);
}
